// CRFLoss_31585189495268
// MI455X (gfx1250) — matrix operand order repaired and checked by reading, not compiled, not run on hardware
//
#include <hip/hip_runtime.h>

typedef __attribute__((ext_vector_type(16))) __bf16 v16bf;
typedef __attribute__((ext_vector_type(8)))  __bf16 v8bf;
typedef __attribute__((ext_vector_type(8)))  float  v8f;

#define CRF_B 256
#define CRF_K 64
#define CRF_T 2048
#define TC    8      // timesteps staged per chunk (also offset refresh period)
#define TILEB 16     // batches per block (WMMA M)
#define LN2F    0.69314718055994531f
#define INVLN2F 1.4426950408889634f

// One block = 16 batches. 4 waves, wave w owns destination states [16w,16w+16).
// alpha kept absolute in f32, log2 domain. P/E in bf16 (f32 dynamic range), so
// the stabilizing offset only needs a per-chunk exact refresh, not per-step.
// Single barrier per scan step (ping-pong P buffer covers the WAR hazard).
__global__ __launch_bounds__(128) void crf_fwd_kernel(
    const int*   __restrict__ label,
    const float* __restrict__ logits,      // (B, K, T)
    const float* __restrict__ start_t,     // (K)
    const float* __restrict__ trans,       // (K, K) row = prev state
    const float* __restrict__ end_t,       // (K)
    float*       __restrict__ ws)          // 16 block partials of sum(llh)
{
  __shared__ __align__(16) float  em_lds[TC * 64 * 16]; // [t][k][m] log2 units
  __shared__ __align__(16) __bf16 p_buf[2][16 * 64];    // ping-pong [m][k]
  __shared__ float maxbuf[4 * 16];                       // [wave][row]
  __shared__ __align__(16) float  fin[16 * 64];          // [m][j]
  __shared__ float redbuf[16];

  const int tid  = threadIdx.x;
  const int w    = tid >> 5;
  const int lane = tid & 31;
  const int col  = lane & 15;
  const int hsel = lane >> 4;
  const int b0   = blockIdx.x * TILEB;
  const int jj   = w * 16 + col;     // destination state column

  // ---- B fragments: E = exp(trans) in bf16, column tile jj.
  // Dense 16-bit B 32x16 layout (matches A): e<8 -> K=e+hsel*8,
  // else K=e+8+hsel*8.
  v16bf bfrag0, bfrag1;
  #pragma unroll
  for (int e = 0; e < 16; ++e) {
    int kb = (e < 8) ? (e + hsel * 8) : (e + 8 + hsel * 8);
    bfrag0[e] = (__bf16)__expf(trans[(kb)      * 64 + jj]);
    bfrag1[e] = (__bf16)__expf(trans[(kb + 32) * 64 + jj]);
  }

  float c[8];          // alpha tile (C layout: row M = v + hsel*8, col jj)
  float off[8];        // per-row stabilizing offset (log2), chunk-constant
  float numAcc = 0.0f;
  int   prevTag = 0;
  int   ping = 0;

  for (int t0 = 0; t0 < CRF_T; t0 += TC) {
    __syncthreads();   // em_lds readers of previous chunk done
    // ---- stage emissions chunk, scaled to log2 units
    #pragma unroll
    for (int r = 0; r < 8; ++r) {
      int R = r * 128 + tid;
      int m = R >> 6, k = R & 63;
      const float* gbase = logits + ((size_t)((b0 + m) * 64 + k)) * CRF_T + t0;
      const float4* gp = (const float4*)gbase;
      float4 v0 = gp[0], v1 = gp[1];
      if (t0 + TC < CRF_T) __builtin_prefetch(gbase + TC, 0, 1);
      em_lds[(0*64 + k)*16 + m] = v0.x * INVLN2F;
      em_lds[(1*64 + k)*16 + m] = v0.y * INVLN2F;
      em_lds[(2*64 + k)*16 + m] = v0.z * INVLN2F;
      em_lds[(3*64 + k)*16 + m] = v0.w * INVLN2F;
      em_lds[(4*64 + k)*16 + m] = v1.x * INVLN2F;
      em_lds[(5*64 + k)*16 + m] = v1.y * INVLN2F;
      em_lds[(6*64 + k)*16 + m] = v1.z * INVLN2F;
      em_lds[(7*64 + k)*16 + m] = v1.w * INVLN2F;
    }
    // ---- per-chunk exact offset refresh: butterfly row max of own slice
    if (t0 > 0) {
      #pragma unroll
      for (int v = 0; v < 8; ++v) {
        float x = c[v];
        x = fmaxf(x, __shfl_xor(x, 1, 32));
        x = fmaxf(x, __shfl_xor(x, 2, 32));
        x = fmaxf(x, __shfl_xor(x, 4, 32));
        x = fmaxf(x, __shfl_xor(x, 8, 32));
        if (col == 0) maxbuf[w*16 + v + hsel*8] = x;
      }
    }
    __syncthreads();   // staging writes + maxbuf visible

    int ttStart = 0;
    if (t0 == 0) {
      const float4* e0 = (const float4*)&em_lds[(0*64 + jj)*16 + hsel*8];
      float4 ea = e0[0], eb = e0[1];
      float sj = start_t[jj] * INVLN2F;
      c[0]=sj+ea.x; c[1]=sj+ea.y; c[2]=sj+ea.z; c[3]=sj+ea.w;
      c[4]=sj+eb.x; c[5]=sj+eb.y; c[6]=sj+eb.z; c[7]=sj+eb.w;
      #pragma unroll
      for (int v = 0; v < 8; ++v) off[v] = 0.f;
      ttStart = 1;
    } else {
      #pragma unroll
      for (int v = 0; v < 8; ++v) {
        int row = v + hsel * 8;
        off[v] = fmaxf(fmaxf(maxbuf[row],      maxbuf[16 + row]),
                       fmaxf(maxbuf[32 + row], maxbuf[48 + row]));
      }
    }

    // ---- numerator gather-sum for this chunk (wave 0, one lane per batch)
    if (w == 0 && lane < 16) {
      int m = lane;
      #pragma unroll
      for (int tt = 0; tt < TC; ++tt) {
        int t   = t0 + tt;
        int tag = label[(size_t)(b0 + m) * CRF_T + t];
        float e = em_lds[(tt*64 + tag)*16 + m] * LN2F;   // back to nats
        if (t == 0) numAcc  = start_t[tag] + e;
        else        numAcc += e + trans[prevTag * 64 + tag];
        prevTag = tag;
      }
    }

    for (int tt = ttStart; tt < TC; ++tt) {
      // ---- write P slice: exp2(alpha - off), bf16, row-major [m][k]
      __bf16* pb = p_buf[ping];
      #pragma unroll
      for (int v = 0; v < 8; ++v) {
        pb[(v + hsel*8) * 64 + jj] =
            (__bf16)__builtin_amdgcn_exp2f(c[v] - off[v]);
      }
      __syncthreads();   // P exchange (also covers iter i+2 WAR via ping-pong)

      // ---- A fragments (16-bit A layout: e<8 -> K=e+hsel*8, else K=e+8+hsel*8)
      int abase = col * 64;
      v8bf lo0 = *(const v8bf*)&pb[abase +      hsel*8];
      v8bf hi0 = *(const v8bf*)&pb[abase + 16 + hsel*8];
      v8bf lo1 = *(const v8bf*)&pb[abase + 32 + hsel*8];
      v8bf hi1 = *(const v8bf*)&pb[abase + 48 + hsel*8];
      v16bf a0, a1;
      #pragma unroll
      for (int e = 0; e < 8; ++e) {
        a0[e] = lo0[e]; a0[e+8] = hi0[e];
        a1[e] = lo1[e]; a1[e+8] = hi1[e];
      }
      v8f acc = {0.f,0.f,0.f,0.f,0.f,0.f,0.f,0.f};
      acc = __builtin_amdgcn_wmma_f32_16x16x32_bf16(
                false, a0, false, bfrag0, (short)0, acc, false, false);
      acc = __builtin_amdgcn_wmma_f32_16x16x32_bf16(
                false, a1, false, bfrag1, (short)0, acc, false, false);

      const float4* e0 = (const float4*)&em_lds[(tt*64 + jj)*16 + hsel*8];
      float4 ea = e0[0], eb = e0[1];
      c[0] = ea.x + off[0] + __builtin_amdgcn_logf(acc[0]);
      c[1] = ea.y + off[1] + __builtin_amdgcn_logf(acc[1]);
      c[2] = ea.z + off[2] + __builtin_amdgcn_logf(acc[2]);
      c[3] = ea.w + off[3] + __builtin_amdgcn_logf(acc[3]);
      c[4] = eb.x + off[4] + __builtin_amdgcn_logf(acc[4]);
      c[5] = eb.y + off[5] + __builtin_amdgcn_logf(acc[5]);
      c[6] = eb.z + off[6] + __builtin_amdgcn_logf(acc[6]);
      c[7] = eb.w + off[7] + __builtin_amdgcn_logf(acc[7]);
      ping ^= 1;
    }
  }

  if (w == 0 && lane < 16) numAcc += end_t[prevTag];

  // ---- denominator: log2-sum-exp2 over states of (alpha + end/ln2)
  #pragma unroll
  for (int v = 0; v < 8; ++v)
    fin[(v + hsel*8) * 64 + jj] = c[v] + end_t[jj] * INVLN2F;
  __syncthreads();

  if (w == 0 && lane < 16) {
    int m = lane;
    float mx = -3.4e38f;
    for (int j = 0; j < 64; ++j) mx = fmaxf(mx, fin[m*64 + j]);
    float s = 0.f;
    for (int j = 0; j < 64; ++j)
      s += __builtin_amdgcn_exp2f(fin[m*64 + j] - mx);
    float denom2 = mx + __builtin_amdgcn_logf(s);   // log2 units
    redbuf[m] = numAcc - LN2F * denom2;
  }
  __syncthreads();
  if (tid == 0) {
    float s = 0.f;
    for (int m = 0; m < 16; ++m) s += redbuf[m];
    ws[blockIdx.x] = s;
  }
}

__global__ void crf_reduce_kernel(const float* __restrict__ ws,
                                  float* __restrict__ out) {
  if (threadIdx.x == 0) {
    float s = 0.f;
    for (int i = 0; i < 16; ++i) s += ws[i];
    out[0] = -s * (1.0f / (float)CRF_B);
  }
}

extern "C" void kernel_launch(void* const* d_in, const int* in_sizes, int n_in,
                              void* d_out, int out_size, void* d_ws, size_t ws_size,
                              hipStream_t stream) {
  const int*   label   = (const int*)  d_in[0];
  const float* logits  = (const float*)d_in[1];
  // d_in[2] = mask (all ones for this problem) -- unused
  const float* start_t = (const float*)d_in[3];
  const float* trans   = (const float*)d_in[4];
  const float* end_t   = (const float*)d_in[5];
  float* out = (float*)d_out;
  float* ws  = (float*)d_ws;

  crf_fwd_kernel<<<CRF_B / TILEB, 128, 0, stream>>>(
      label, logits, start_t, trans, end_t, ws);
  crf_reduce_kernel<<<1, 32, 0, stream>>>(ws, out);
}
